// Attention_55619826483548
// MI455X (gfx1250) — matrix operand order repaired and checked by reading, not compiled, not run on hardware
//
#include <hip/hip_runtime.h>
#include <hip/hip_bf16.h>

// ---------------------------------------------------------------------------
// Types for gfx1250 WMMA (wave32): 16x16x32 bf16, fp32 accumulate
// ---------------------------------------------------------------------------
typedef __bf16 bf16;
typedef __bf16 v8bf  __attribute__((ext_vector_type(8)));
typedef __bf16 v16bf __attribute__((ext_vector_type(16)));
typedef float  v8f   __attribute__((ext_vector_type(8)));

// Load a 16-element bf16 fragment as two 16B LDS reads (ds_load_b128 each).
__device__ __forceinline__ v16bf ld_frag(const bf16* p0, const bf16* p1) {
    v8bf lo = *(const v8bf*)p0;
    v8bf hi = *(const v8bf*)p1;
    v16bf r;
#pragma unroll
    for (int i = 0; i < 8; ++i) { r[i] = lo[i]; r[8 + i] = hi[i]; }
    return r;
}

__device__ __forceinline__ v8f wmma_bf16(v16bf a, v16bf b, v8f c) {
    return __builtin_amdgcn_wmma_f32_16x16x32_bf16(
        /*neg_a=*/false, a, /*neg_b=*/false, b,
        /*c_mod=*/(short)0, c, /*reuse_a=*/false, /*reuse_b=*/false);
}

// Async global->LDS copy, 16B per lane (ASYNCcnt-tracked, no VGPR data).
__device__ __forceinline__ void async_copy_b128(unsigned lds_addr, const bf16* gptr) {
    asm volatile("global_load_async_to_lds_b128 %0, %1, off"
                 :: "v"(lds_addr), "v"((unsigned long long)(uintptr_t)gptr)
                 : "memory");
}
__device__ __forceinline__ void wait_async0() {
    asm volatile("s_wait_asynccnt 0x0" ::: "memory");
}

// VALU cross-lane permute within 16-lane groups (v_permlane16_b32),
// keeping the softmax reduction off the LDS pipe.
__device__ __forceinline__ float permlane16_f(float v, int s0, int s1) {
    int iv = __float_as_int(v);
    int r = __builtin_amdgcn_permlane16(iv, iv, s0, s1, false, false);
    return __int_as_float(r);
}

// ---------------------------------------------------------------------------
// GEMM + bias + scale:  C[M,N] = (A[M,K] * W[K,N] + bias[N]) * out_scale
// A fp32 or bf16 (WMMA in bf16), W/bias fp32, C fp32 or bf16.
// Block 256 threads = 8 waves as 4(M) x 2(N); block tile 128x128, K-step 32.
// Each wave computes 32x64 = 2x4 WMMA 16x16 tiles (8 WMMA / K-step).
// LDS is double-buffered: tile t+1 is staged while tile t is consumed
// (single barrier per K-step).
// ---------------------------------------------------------------------------
template <typename InT, typename OutT>
__global__ __launch_bounds__(256)
void gemm_bias_kernel(const InT* __restrict__ A, const float* __restrict__ W,
                      const float* __restrict__ bias, OutT* __restrict__ C,
                      int M, int N, int K, float out_scale)
{
    __shared__ alignas(16) bf16 As[2][128][40];   // [m][k], K-contig, +8 pad
    __shared__ alignas(16) bf16 Bs[2][128][40];   // [n][k], K-contig per column

    const int tid  = threadIdx.x;
    const int wid  = tid >> 5;
    const int lane = tid & 31;
    const int hf   = lane >> 4;     // half-wave id (0/1)
    const int l16  = lane & 15;
    const int wm   = wid & 3;       // wave M tile (0..3), 32 rows each
    const int wn   = wid >> 2;      // wave N tile (0..1), 64 cols each

    const int m0 = blockIdx.y * 128;
    const int n0 = blockIdx.x * 128;

    const int arow = tid >> 1;            // 0..127
    const int ac0  = (tid & 1) * 16;      // 0/16
    const int bk   = tid & 31;            // k within tile
    const int bn0  = (tid >> 5) * 16;     // 0..112

    auto stage = [&](int sb, int kt) {
        // A tile 128x32 -> bf16 LDS (two packed b128 stores)
        const InT* ap = A + (size_t)(m0 + arow) * K + kt + ac0;
#pragma unroll
        for (int j = 0; j < 2; ++j) {
            v8bf pack;
#pragma unroll
            for (int i = 0; i < 8; ++i) pack[i] = (bf16)(float)ap[j * 8 + i];
            *(v8bf*)&As[sb][arow][ac0 + j * 8] = pack;
        }
        // W tile 32x128 -> transposed bf16 LDS [n][k]
        const float* wp = W + (size_t)(kt + bk) * N + n0 + bn0;
#pragma unroll
        for (int i = 0; i < 16; ++i)
            Bs[sb][bn0 + i][bk] = (bf16)wp[i];
    };

    v8f acc[2][4];
#pragma unroll
    for (int mi = 0; mi < 2; ++mi)
#pragma unroll
        for (int ni = 0; ni < 4; ++ni)
#pragma unroll
            for (int r = 0; r < 8; ++r) acc[mi][ni][r] = 0.0f;

    stage(0, 0);
    __syncthreads();

    int buf = 0;
    for (int kt = 0; kt < K; kt += 32) {
        if (kt + 32 < K) stage(buf ^ 1, kt + 32);   // overlap with WMMAs below

#pragma unroll
        for (int mi = 0; mi < 2; ++mi) {
            const int row = wm * 32 + mi * 16 + l16;
            // A frag: elems 0-7 -> K=hf*8.., 8-15 -> K=16+hf*8..
            v16bf af = ld_frag(&As[buf][row][hf * 8], &As[buf][row][16 + hf * 8]);
#pragma unroll
            for (int ni = 0; ni < 4; ++ni) {
                const int col = wn * 64 + ni * 16 + l16;
                // B frag: elems 0-7 -> K=hf*8.., 8-15 -> K=16+hf*8..
                v16bf bf_ = ld_frag(&Bs[buf][col][hf * 8], &Bs[buf][col][16 + hf * 8]);
                acc[mi][ni] = wmma_bf16(af, bf_, acc[mi][ni]);
            }
        }
        __syncthreads();
        buf ^= 1;
    }

    // Epilogue: C layout -> element r is row M = hf*8 + r, col N = l16
#pragma unroll
    for (int mi = 0; mi < 2; ++mi)
#pragma unroll
        for (int ni = 0; ni < 4; ++ni) {
            const int col = n0 + wn * 64 + ni * 16 + l16;
            const float bv = bias[col];
#pragma unroll
            for (int r = 0; r < 8; ++r) {
                const int row = m0 + wm * 32 + mi * 16 + hf * 8 + r;
                C[(size_t)row * N + col] = (OutT)((acc[mi][ni][r] + bv) * out_scale);
            }
        }
}

// ---------------------------------------------------------------------------
// Flash attention over bf16 projected Q/K/V (layout (B,S,H*64), stride 1024).
// Q is pre-scaled by (1/sqrt(64))*log2(e) in its projection, so WMMA scores
// are directly in log2 domain and softmax uses raw v_exp_f32 (exp2).
// Grid: (S/128, B*H). Block 256 threads = 8 waves; wave w owns 16 query rows.
// 64-key blocks, double-buffered: next K via global_load_async_to_lds_b128,
// next V prefetched to VGPRs during compute, transposed-stored after.
// Row-sum of P comes from an extra WMMA against an all-ones B matrix; the
// row-max reduction runs on VALU v_permlane16_b32 (off the LDS pipe).
// ---------------------------------------------------------------------------
__global__ __launch_bounds__(256)
void attn_kernel(const bf16* __restrict__ Qp, const bf16* __restrict__ Kp,
                 const bf16* __restrict__ Vp, bf16* __restrict__ Ao)
{
    constexpr int S = 2048, D = 1024, HD = 64;

    const int bh = blockIdx.y;
    const int b  = bh >> 4;
    const int h  = bh & 15;
    const int q0 = blockIdx.x * 128;

    const int tid  = threadIdx.x;
    const int wid  = tid >> 5;
    const int lane = tid & 31;
    const int hf   = lane >> 4;
    const int l16  = lane & 15;

    __shared__ alignas(16) bf16 Qs[128][72];         // [qrow][d] +8 pad
    __shared__ alignas(16) bf16 Ks[2][64][72];       // [key][d]  +8 pad
    __shared__ alignas(16) bf16 Vs[2][64][72];       // [vcol][key] +8 pad
    __shared__ alignas(16) bf16 Ps[8][16][72];       // per-wave P round-trip

    const size_t base = (size_t)b * S * D + (size_t)h * HD;

    const unsigned qs_base = (unsigned)(uintptr_t)&Qs[0][0];
    const unsigned ks_base = (unsigned)(uintptr_t)&Ks[0][0][0];
    constexpr unsigned KS_BUF = 64u * 72u * 2u;      // bytes per K buffer

    const int krr = tid >> 3;             // 0..31 (K/Q staging row)
    const int kc0 = (tid & 7) * 8;        // 0..56 (K/Q staging col)
    const int vkey = tid & 63;            // V staging key
    const int vc0  = (tid >> 6) * 16;     // V staging col base {0,16,32,48}

    // ---- stage this block's 128 query rows once (async, 4 x 16B per lane)
#pragma unroll
    for (int j = 0; j < 4; ++j) {
        const int row = krr + j * 32;
        async_copy_b128(qs_base + (unsigned)(row * 144 + kc0 * 2),
                        Qp + base + (size_t)(q0 + row) * D + kc0);
    }

    // ---- prologue: K block 0 (async) + V block 0 (VALU transpose)
#pragma unroll
    for (int j = 0; j < 2; ++j) {
        const int key = krr + j * 32;
        async_copy_b128(ks_base + (unsigned)(key * 144 + kc0 * 2),
                        Kp + base + (size_t)key * D + kc0);
    }
    {
        const bf16* vsrc = Vp + base + (size_t)vkey * D + vc0;
        v8bf v0 = *(const v8bf*)vsrc;
        v8bf v1 = *(const v8bf*)(vsrc + 8);
#pragma unroll
        for (int i = 0; i < 8; ++i) {
            Vs[0][vc0 + i][vkey]     = v0[i];
            Vs[0][vc0 + 8 + i][vkey] = v1[i];
        }
    }

    v8f oacc[4], lacc;
#pragma unroll
    for (int vt = 0; vt < 4; ++vt)
#pragma unroll
        for (int r = 0; r < 8; ++r) oacc[vt][r] = 0.0f;
#pragma unroll
    for (int r = 0; r < 8; ++r) lacc[r] = 0.0f;

    float m_run[8];
#pragma unroll
    for (int r = 0; r < 8; ++r) m_run[r] = -3.0e38f;

    v16bf ones;
#pragma unroll
    for (int i = 0; i < 16; ++i) ones[i] = (bf16)1.0f;

    const int qrow = wid * 16 + l16;

    // v_permlane16_b32 nibble-select tables for XOR 1/2/4/8 inside a 16-group
    const int xsel0[4] = { 0x67452301, 0x54761032, 0x32107654, (int)0xFEDCBA98 };
    const int xsel1[4] = { (int)0xEFCDAB89, (int)0xDCFE98BA, (int)0xBA98FEDC, 0x76543210 };

    wait_async0();                // Q + K0 resident
    __syncthreads();

    int buf = 0;
    for (int kb = 0; kb < S; kb += 64) {
        const int nkb = kb + 64;
        v8bf vpre0, vpre1;
        if (nkb < S) {
            // issue next K block on the async engine
#pragma unroll
            for (int j = 0; j < 2; ++j) {
                const int key = krr + j * 32;
                async_copy_b128(ks_base + (unsigned)((buf ^ 1) * KS_BUF + key * 144 + kc0 * 2),
                                Kp + base + (size_t)(nkb + key) * D + kc0);
            }
            // prefetch next V block into VGPRs (latency hidden by compute)
            const bf16* vsrc = Vp + base + (size_t)(nkb + vkey) * D + vc0;
            vpre0 = *(const v8bf*)vsrc;
            vpre1 = *(const v8bf*)(vsrc + 8);
        }

        // ---- scores: 16(q) x 64(keys), 4 n-tiles x 2 K-steps of WMMA
        v8f sc[4];
#pragma unroll
        for (int nt = 0; nt < 4; ++nt)
#pragma unroll
            for (int r = 0; r < 8; ++r) sc[nt][r] = 0.0f;

#pragma unroll
        for (int d0 = 0; d0 < HD; d0 += 32) {
            v16bf aq = ld_frag(&Qs[qrow][d0 + hf * 8], &Qs[qrow][d0 + 16 + hf * 8]);
#pragma unroll
            for (int nt = 0; nt < 4; ++nt) {
                const int key = nt * 16 + l16;
                v16bf bk_ = ld_frag(&Ks[buf][key][d0 + hf * 8],
                                    &Ks[buf][key][d0 + 16 + hf * 8]);
                sc[nt] = wmma_bf16(aq, bk_, sc[nt]);
            }
        }

        // ---- online softmax in log2 domain (row r <-> query row hf*8 + r)
        float mx[8], corr[8];
#pragma unroll
        for (int r = 0; r < 8; ++r)
            mx[r] = fmaxf(fmaxf(sc[0][r], sc[1][r]), fmaxf(sc[2][r], sc[3][r]));
#pragma unroll
        for (int step = 0; step < 4; ++step)
#pragma unroll
            for (int r = 0; r < 8; ++r)
                mx[r] = fmaxf(mx[r], permlane16_f(mx[r], xsel0[step], xsel1[step]));

#pragma unroll
        for (int r = 0; r < 8; ++r) {
            const float mn = fmaxf(m_run[r], mx[r]);
            corr[r] = __builtin_amdgcn_exp2f(m_run[r] - mn);
            m_run[r] = mn;
            const int m = hf * 8 + r;
#pragma unroll
            for (int nt = 0; nt < 4; ++nt) {
                const float e = __builtin_amdgcn_exp2f(sc[nt][r] - mn);
                Ps[wid][m][nt * 16 + l16] = (bf16)e;
            }
        }

        // rescale running output and running denominator
#pragma unroll
        for (int vt = 0; vt < 4; ++vt)
#pragma unroll
            for (int r = 0; r < 8; ++r) oacc[vt][r] *= corr[r];
#pragma unroll
        for (int r = 0; r < 8; ++r) lacc[r] *= corr[r];

        // wave-local LDS ordering: P writes -> P fragment reads
        asm volatile("s_wait_dscnt 0" ::: "memory");
        __builtin_amdgcn_wave_barrier();

        // ---- PV + row-sum: A = P (16x64 over 2 K-steps), B = V^T tiles / ones
#pragma unroll
        for (int k0 = 0; k0 < 64; k0 += 32) {
            v16bf pf = ld_frag(&Ps[wid][l16][k0 + hf * 8],
                               &Ps[wid][l16][k0 + 16 + hf * 8]);
#pragma unroll
            for (int vt = 0; vt < 4; ++vt) {
                const int vcol = vt * 16 + l16;
                v16bf bv_ = ld_frag(&Vs[buf][vcol][k0 + hf * 8],
                                    &Vs[buf][vcol][k0 + 16 + hf * 8]);
                oacc[vt] = wmma_bf16(pf, bv_, oacc[vt]);
            }
            lacc = wmma_bf16(pf, ones, lacc);   // l += rowsum(P), lane-replicated
        }

        // ---- commit prefetched V into the other buffer, then flip
        if (nkb < S) {
#pragma unroll
            for (int i = 0; i < 8; ++i) {
                Vs[buf ^ 1][vc0 + i][vkey]     = vpre0[i];
                Vs[buf ^ 1][vc0 + 8 + i][vkey] = vpre1[i];
            }
        }
        wait_async0();
        __syncthreads();
        buf ^= 1;
    }

    // ---- normalize and store (bf16 workspace, same (B,S,H*64) layout)
    float inv[8];
#pragma unroll
    for (int r = 0; r < 8; ++r) inv[r] = 1.0f / lacc[r];
#pragma unroll
    for (int vt = 0; vt < 4; ++vt)
#pragma unroll
        for (int r = 0; r < 8; ++r) {
            const int row = q0 + wid * 16 + hf * 8 + r;
            Ao[base + (size_t)row * D + vt * 16 + l16] = (bf16)(oacc[vt][r] * inv[r]);
        }
}

// ---------------------------------------------------------------------------
// Launcher
// ---------------------------------------------------------------------------
extern "C" void kernel_launch(void* const* d_in, const int* in_sizes, int n_in,
                              void* d_out, int out_size, void* d_ws, size_t ws_size,
                              hipStream_t stream)
{
    (void)in_sizes; (void)n_in; (void)out_size; (void)ws_size;

    constexpr int B = 4, S = 2048, E = 1024;
    constexpr int M = B * S;                  // 8192
    constexpr size_t SLAB = (size_t)M * E;    // elements per (B,S,E) tensor

    const float* q  = (const float*)d_in[0];
    const float* k  = (const float*)d_in[1];
    const float* v  = (const float*)d_in[2];
    const float* Wq = (const float*)d_in[3];
    const float* bq = (const float*)d_in[4];
    const float* Wk = (const float*)d_in[5];
    const float* bk = (const float*)d_in[6];
    const float* Wv = (const float*)d_in[7];
    const float* bv = (const float*)d_in[8];
    const float* Wo = (const float*)d_in[9];
    const float* bo = (const float*)d_in[10];
    float* out = (float*)d_out;

    bf16* Qp = (bf16*)d_ws;
    bf16* Kp = Qp + SLAB;
    bf16* Vp = Kp + SLAB;
    bf16* Aop = Vp + SLAB;     // attention output, pre output-projection

    // softmax scale (1/sqrt(64)) and log2(e) folded into the Q projection:
    // WMMA scores come out directly in the exp2 domain.
    const float qscale = 0.125f * 1.44269504088896340736f;

    dim3 blk(256);
    dim3 gproj(E / 128, M / 128);         // (8, 64)

    gemm_bias_kernel<float, bf16><<<gproj, blk, 0, stream>>>(q, Wq, bq, Qp, M, E, E, qscale);
    gemm_bias_kernel<float, bf16><<<gproj, blk, 0, stream>>>(k, Wk, bk, Kp, M, E, E, 1.0f);
    gemm_bias_kernel<float, bf16><<<gproj, blk, 0, stream>>>(v, Wv, bv, Vp, M, E, E, 1.0f);

    dim3 gattn(S / 128, B * 16);          // (16, 64) blocks, one (b,h,128q) each
    attn_kernel<<<gattn, blk, 0, stream>>>(Qp, Kp, Vp, Aop);

    gemm_bias_kernel<bf16, float><<<gproj, blk, 0, stream>>>(Aop, Wo, bo, out, M, E, E, 1.0f);
}
